// Qwen3OmniMoeThinkerTextAttention_29961691857443
// MI455X (gfx1250) — compile-verified, matrix operand order checked by reading, not run on hardware
//
#include <hip/hip_runtime.h>

// ---------------------------------------------------------------- constants
constexpr int B   = 2;
constexpr int S   = 2048;
constexpr int HID = 2048;
constexpr int H   = 32;
constexpr int KV  = 4;
constexpr int D   = 128;
constexpr int G   = H / KV;           // 8
constexpr float EPS   = 1e-6f;
constexpr float SCALE = 0.08838834764831845f;   // 128^-0.5

// ---------------------------------------------------------------- types
typedef __bf16 bf16_t;
typedef __attribute__((ext_vector_type(16))) __bf16 v16bf;
typedef __attribute__((ext_vector_type(8)))  float  v8f;

union FragA {                 // 16x32 bf16 A/B fragment = 32 bytes/lane
    uint4 q[2];
    v16bf bf;
};

__device__ inline unsigned short f2bf(float f) {
    unsigned u = __float_as_uint(f);
    u += 0x7FFFu + ((u >> 16) & 1u);      // round-to-nearest-even
    return (unsigned short)(u >> 16);
}

__device__ inline v8f vzero8() {
    v8f z = {0.f, 0.f, 0.f, 0.f, 0.f, 0.f, 0.f, 0.f};
    return z;
}

__device__ inline v8f wmma_bf16(const v16bf& a, const v16bf& b, const v8f& c) {
    return __builtin_amdgcn_wmma_f32_16x16x32_bf16(
        /*neg_a=*/false, a, /*neg_b=*/false, b,
        /*c_mod=*/(short)0, c, /*reuse_a=*/false, /*reuse_b=*/false);
}

// ---- CDNA5 async global -> LDS copy (ASYNCcnt tracked), 16B per lane
__device__ inline void async_copy_b128(const void* gptr, void* lptr) {
    // generic shared address: low 32 bits are the LDS byte offset
    unsigned lds_off = (unsigned)(uintptr_t)lptr;
    asm volatile("global_load_async_to_lds_b128 %0, %1, off"
                 :: "v"(lds_off), "v"(gptr) : "memory");
}
__device__ inline void wait_async0() {
    asm volatile("s_wait_asynccnt 0x0" ::: "memory");
}

// ================================================================ GEMM
// C[M,N] = A[M,K] * W[N,K]^T.  A fp32 or bf16 (template), W fp32 weights,
// C fp32 or bf16 (template).  M,N multiples of 128; K multiple of 32.
// Block: 256 threads = 8 waves arranged 4(m) x 2(n); wave tile 32x64.
template <bool ABF16, bool CBF16>
__global__ __launch_bounds__(256)
void gemm_wmma(const void* __restrict__ Aptr, const float* __restrict__ Wp,
               void* __restrict__ Cptr, int M, int N, int Kd) {
    constexpr int LDT = 40;                       // padded 32-elem row
    __shared__ __attribute__((aligned(16))) unsigned short lA[128 * LDT];
    __shared__ __attribute__((aligned(16))) unsigned short lB[128 * LDT];

    const int tid  = threadIdx.x;
    const int wid  = tid >> 5;
    const int lane = tid & 31;
    const int half = lane >> 4;
    const int l16  = lane & 15;

    const int m0 = blockIdx.y * 128;
    const int n0 = blockIdx.x * 128;
    const int wm = (wid >> 1) * 32;               // 0,32,64,96
    const int wn = (wid & 1) * 64;                // 0,64

    v8f acc[2][4];
#pragma unroll
    for (int i = 0; i < 2; ++i)
#pragma unroll
        for (int j = 0; j < 4; ++j) acc[i][j] = vzero8();

    for (int k0 = 0; k0 < Kd; k0 += 32) {
        // ---- stage A tile (128 x 32) into LDS as bf16
        if (ABF16) {
            const unsigned short* A = (const unsigned short*)Aptr;
#pragma unroll
            for (int i = 0; i < 2; ++i) {
                int idx = tid * 2 + i;            // 0..511
                int r = idx >> 2, seg = idx & 3;
                async_copy_b128(A + (size_t)(m0 + r) * Kd + k0 + seg * 8,
                                &lA[r * LDT + seg * 8]);
            }
        } else {
            const float* A = (const float*)Aptr;
#pragma unroll
            for (int i = 0; i < 4; ++i) {
                int idx = tid * 4 + i;            // 0..1023
                int r = idx >> 3, seg = idx & 7;
                const float* src = A + (size_t)(m0 + r) * Kd + k0 + seg * 4;
                if (k0 + 32 < Kd) __builtin_prefetch(src + 32, 0, 0);
                float4 v = *(const float4*)src;
                unsigned short* dst = &lA[r * LDT + seg * 4];
                dst[0] = f2bf(v.x); dst[1] = f2bf(v.y);
                dst[2] = f2bf(v.z); dst[3] = f2bf(v.w);
            }
        }
        // ---- stage W tile (128 x 32) into LDS as bf16
#pragma unroll
        for (int i = 0; i < 4; ++i) {
            int idx = tid * 4 + i;
            int r = idx >> 3, seg = idx & 7;
            const float* src = Wp + (size_t)(n0 + r) * Kd + k0 + seg * 4;
            if (k0 + 32 < Kd) __builtin_prefetch(src + 32, 0, 0);
            float4 v = *(const float4*)src;
            unsigned short* dst = &lB[r * LDT + seg * 4];
            dst[0] = f2bf(v.x); dst[1] = f2bf(v.y);
            dst[2] = f2bf(v.z); dst[3] = f2bf(v.w);
        }
        if (ABF16) wait_async0();                 // ASYNCcnt -> 0 before barrier
        __syncthreads();

        // ---- fragments + 8 WMMAs
        FragA a[2], b[4];
#pragma unroll
        for (int i = 0; i < 2; ++i) {
            const unsigned short* p = &lA[(wm + i * 16 + l16) * LDT + half * 8];
            a[i].q[0] = *(const uint4*)p;
            a[i].q[1] = *(const uint4*)(p + 16);
        }
#pragma unroll
        for (int j = 0; j < 4; ++j) {
            const unsigned short* p = &lB[(wn + j * 16 + l16) * LDT + half * 8];
            b[j].q[0] = *(const uint4*)p;
            b[j].q[1] = *(const uint4*)(p + 16);
        }
#pragma unroll
        for (int i = 0; i < 2; ++i)
#pragma unroll
            for (int j = 0; j < 4; ++j)
                acc[i][j] = wmma_bf16(a[i].bf, b[j].bf, acc[i][j]);
        __syncthreads();
    }

    // ---- epilogue: C layout row = vgpr + 8*half, col = l16
#pragma unroll
    for (int i = 0; i < 2; ++i)
#pragma unroll
        for (int j = 0; j < 4; ++j) {
            int col = n0 + wn + j * 16 + l16;
#pragma unroll
            for (int e = 0; e < 8; ++e) {
                int row = m0 + wm + i * 16 + half * 8 + e;
                float v = acc[i][j][e];
                if (CBF16)
                    ((unsigned short*)Cptr)[(size_t)row * N + col] = f2bf(v);
                else
                    ((float*)Cptr)[(size_t)row * N + col] = v;
            }
        }
}

// ================================================================ RMSNorm + RoPE
// One block per token; wave w handles q-heads {w, w+4, ...} and k-head w.
__global__ __launch_bounds__(128)
void norm_rope(const float* __restrict__ qf, const float* __restrict__ kf,
               const float* __restrict__ cosb, const float* __restrict__ sinb,
               const float* __restrict__ qw, const float* __restrict__ kw,
               unsigned short* __restrict__ qb, unsigned short* __restrict__ kb) {
    const int token = blockIdx.x;                 // b*S + s
    const int wid  = threadIdx.x >> 5;
    const int lane = threadIdx.x & 31;
    const int d0   = lane * 4;                    // this lane's 4 dims
    const float sgn = (d0 < 64) ? -1.f : 1.f;     // rotate_half sign

    const float4 cs = *(const float4*)(cosb + (size_t)token * D + d0);
    const float4 sn = *(const float4*)(sinb + (size_t)token * D + d0);

    // ---- queries
    const float4 wq4 = *(const float4*)(qw + d0);
    for (int h = wid; h < H; h += 4) {
        const float4 x = *(const float4*)(qf + ((size_t)token * H + h) * D + d0);
        float ss = x.x * x.x + x.y * x.y + x.z * x.z + x.w * x.w;
#pragma unroll
        for (int m = 1; m < 32; m <<= 1) ss += __shfl_xor(ss, m, 32);
        const float r = rsqrtf(ss * (1.f / 128.f) + EPS);
        float4 xn;
        xn.x = x.x * r * wq4.x; xn.y = x.y * r * wq4.y;
        xn.z = x.z * r * wq4.z; xn.w = x.w * r * wq4.w;
        float4 xo;
        xo.x = __shfl_xor(xn.x, 16, 32); xo.y = __shfl_xor(xn.y, 16, 32);
        xo.z = __shfl_xor(xn.z, 16, 32); xo.w = __shfl_xor(xn.w, 16, 32);
        unsigned short* dst = qb + ((size_t)token * H + h) * D + d0;
        dst[0] = f2bf(xn.x * cs.x + sgn * xo.x * sn.x);
        dst[1] = f2bf(xn.y * cs.y + sgn * xo.y * sn.y);
        dst[2] = f2bf(xn.z * cs.z + sgn * xo.z * sn.z);
        dst[3] = f2bf(xn.w * cs.w + sgn * xo.w * sn.w);
    }

    // ---- keys (KV == 4 waves)
    {
        const int h = wid;
        const float4 wk4 = *(const float4*)(kw + d0);
        const float4 x = *(const float4*)(kf + ((size_t)token * KV + h) * D + d0);
        float ss = x.x * x.x + x.y * x.y + x.z * x.z + x.w * x.w;
#pragma unroll
        for (int m = 1; m < 32; m <<= 1) ss += __shfl_xor(ss, m, 32);
        const float r = rsqrtf(ss * (1.f / 128.f) + EPS);
        float4 xn;
        xn.x = x.x * r * wk4.x; xn.y = x.y * r * wk4.y;
        xn.z = x.z * r * wk4.z; xn.w = x.w * r * wk4.w;
        float4 xo;
        xo.x = __shfl_xor(xn.x, 16, 32); xo.y = __shfl_xor(xn.y, 16, 32);
        xo.z = __shfl_xor(xn.z, 16, 32); xo.w = __shfl_xor(xn.w, 16, 32);
        unsigned short* dst = kb + ((size_t)token * KV + h) * D + d0;
        dst[0] = f2bf(xn.x * cs.x + sgn * xo.x * sn.x);
        dst[1] = f2bf(xn.y * cs.y + sgn * xo.y * sn.y);
        dst[2] = f2bf(xn.z * cs.z + sgn * xo.z * sn.z);
        dst[3] = f2bf(xn.w * cs.w + sgn * xo.w * sn.w);
    }
}

// ================================================================ Flash attention
// Block = 128 threads (4 waves); each wave owns 16 query rows; block covers 64.
// grid = (B*H, S/64). Causal, GQA (kv head = h / G). Online softmax.
__global__ __launch_bounds__(128)
void flash_attn(const unsigned short* __restrict__ qb,
                const unsigned short* __restrict__ kb,
                const unsigned short* __restrict__ vb,
                unsigned short* __restrict__ attnb) {
    constexpr int KLD = 136;   // padded 128-elem K rows
    constexpr int VLD = 40;    // padded 32-elem Vt / P rows
    __shared__ __attribute__((aligned(16))) unsigned short Kt[32 * KLD];   // [n][d]
    __shared__ __attribute__((aligned(16))) unsigned short Vt[128 * VLD];  // [d][n]
    __shared__ __attribute__((aligned(16))) unsigned short Pt[4][16 * VLD];

    const int bh  = blockIdx.x;
    const int b   = bh / H, h = bh % H;
    const int kvh = h / G;
    const int m0b = blockIdx.y * 64;

    const int tid  = threadIdx.x;
    const int wid  = tid >> 5;
    const int lane = tid & 31;
    const int half = lane >> 4;
    const int l16  = lane & 15;
    const int m0   = m0b + wid * 16;

    // ---- Q fragments, 4 k-chunks of 32 (A-matrix layout: row = l16)
    FragA qa[4];
    {
        const unsigned short* qrow =
            qb + (((size_t)(b * S) + m0 + l16) * H + h) * D;
#pragma unroll
        for (int c = 0; c < 4; ++c) {
            qa[c].q[0] = *(const uint4*)(qrow + c * 32 + half * 8);
            qa[c].q[1] = *(const uint4*)(qrow + c * 32 + 16 + half * 8);
        }
    }

    v8f o[8];
#pragma unroll
    for (int ch = 0; ch < 8; ++ch) o[ch] = vzero8();
    float mrow[8], lrow[8];
#pragma unroll
    for (int e = 0; e < 8; ++e) { mrow[e] = -1.0e30f; lrow[e] = 0.f; }

    for (int n0 = 0; n0 < m0b + 64; n0 += 32) {
        // ---- stage K tile [32][128] via async global->LDS (ASYNCcnt)
#pragma unroll
        for (int i = 0; i < 4; ++i) {
            int idx = tid * 4 + i;                 // 0..511
            int r = idx >> 4, seg = idx & 15;
            async_copy_b128(
                kb + (((size_t)(b * S) + n0 + r) * KV + kvh) * D + seg * 8,
                &Kt[r * KLD + seg * 8]);
        }
        // ---- stage V tile transposed -> Vt[d][n] (needs in-flight transpose)
#pragma unroll
        for (int i = 0; i < 4; ++i) {
            int idx = tid * 4 + i;
            int r = idx >> 4, seg = idx & 15;
            uint4 v = *(const uint4*)(
                vb + (((size_t)(b * S) + n0 + r) * KV + kvh) * D + seg * 8);
            const unsigned short* e = (const unsigned short*)&v;
#pragma unroll
            for (int j = 0; j < 8; ++j) Vt[(seg * 8 + j) * VLD + r] = e[j];
        }
        wait_async0();                            // K tile resident in LDS
        __syncthreads();

        // ---- S = Q K^T (two 16-col chunks), scale + causal mask
        v8f sc[2];
#pragma unroll
        for (int nc = 0; nc < 2; ++nc) {
            v8f s = vzero8();
#pragma unroll
            for (int c = 0; c < 4; ++c) {
                FragA kf;
                const unsigned short* p =
                    &Kt[(nc * 16 + l16) * KLD + c * 32 + half * 8];
                kf.q[0] = *(const uint4*)p;
                kf.q[1] = *(const uint4*)(p + 16);
                s = wmma_bf16(qa[c].bf, kf.bf, s);
            }
            int col = n0 + nc * 16 + l16;
#pragma unroll
            for (int e = 0; e < 8; ++e) {
                int row = m0 + half * 8 + e;
                float v = s[e] * SCALE;
                if (col > row) v = -3.0e38f;
                s[e] = v;
            }
            sc[nc] = s;
        }

        // ---- online softmax bookkeeping (row r lives in vgpr e of this half)
        float mnew[8];
#pragma unroll
        for (int e = 0; e < 8; ++e) {
            float v = fmaxf(sc[0][e], sc[1][e]);
#pragma unroll
            for (int msk = 1; msk < 16; msk <<= 1)
                v = fmaxf(v, __shfl_xor(v, msk, 32));
            mnew[e] = fmaxf(mrow[e], v);
        }
        float rs[8];
#pragma unroll
        for (int e = 0; e < 8; ++e) rs[e] = 0.f;
#pragma unroll
        for (int nc = 0; nc < 2; ++nc)
#pragma unroll
            for (int e = 0; e < 8; ++e) {
                float p = __expf(sc[nc][e] - mnew[e]);
                rs[e] += p;
                Pt[wid][(half * 8 + e) * VLD + nc * 16 + l16] = f2bf(p);
            }
        float corr[8];
#pragma unroll
        for (int e = 0; e < 8; ++e) {
#pragma unroll
            for (int msk = 1; msk < 16; msk <<= 1)
                rs[e] += __shfl_xor(rs[e], msk, 32);
            corr[e] = __expf(mrow[e] - mnew[e]);
            lrow[e] = lrow[e] * corr[e] + rs[e];
            mrow[e] = mnew[e];
        }
#pragma unroll
        for (int ch = 0; ch < 8; ++ch)
#pragma unroll
            for (int e = 0; e < 8; ++e) o[ch][e] *= corr[e];

        // ---- O += P V  (A = P from per-wave LDS, B = Vt rows)
        FragA pa;
        {
            const unsigned short* pp = &Pt[wid][l16 * VLD + half * 8];
            pa.q[0] = *(const uint4*)pp;
            pa.q[1] = *(const uint4*)(pp + 16);
        }
#pragma unroll
        for (int ch = 0; ch < 8; ++ch) {
            FragA vf;
            const unsigned short* vp = &Vt[(ch * 16 + l16) * VLD + half * 8];
            vf.q[0] = *(const uint4*)vp;
            vf.q[1] = *(const uint4*)(vp + 16);
            o[ch] = wmma_bf16(pa.bf, vf.bf, o[ch]);
        }
        __syncthreads();
    }

    // ---- normalize + store bf16 (attn[b][s][h][d])
#pragma unroll
    for (int ch = 0; ch < 8; ++ch) {
        int d = ch * 16 + l16;
#pragma unroll
        for (int e = 0; e < 8; ++e) {
            int row = m0 + half * 8 + e;
            float v = o[ch][e] / lrow[e];
            attnb[(((size_t)(b * S) + row) * H + h) * D + d] = f2bf(v);
        }
    }
}

// ================================================================ launch
extern "C" void kernel_launch(void* const* d_in, const int* in_sizes, int n_in,
                              void* d_out, int out_size, void* d_ws, size_t ws_size,
                              hipStream_t stream) {
    const float* x    = (const float*)d_in[0];
    const float* cosb = (const float*)d_in[1];
    const float* sinb = (const float*)d_in[2];
    const float* wq   = (const float*)d_in[3];
    const float* wk   = (const float*)d_in[4];
    const float* wv   = (const float*)d_in[5];
    const float* wo   = (const float*)d_in[6];
    const float* qnw  = (const float*)d_in[7];
    const float* knw  = (const float*)d_in[8];

    char* w = (char*)d_ws;
    float* qf = (float*)w;              w += (size_t)B * S * H  * D * 4;  // 64 MB
    float* kf = (float*)w;              w += (size_t)B * S * KV * D * 4;  //  8 MB
    unsigned short* qbh   = (unsigned short*)w; w += (size_t)B * S * H  * D * 2; // 32 MB
    unsigned short* kbh   = (unsigned short*)w; w += (size_t)B * S * KV * D * 2; //  4 MB
    unsigned short* vbh   = (unsigned short*)w; w += (size_t)B * S * KV * D * 2; //  4 MB
    unsigned short* attnb = (unsigned short*)w;                                  // 32 MB

    const dim3 blk(256);
    // Q/K/V projections
    gemm_wmma<false, false><<<dim3((H * D) / 128, (B * S) / 128), blk, 0, stream>>>(
        x, wq, qf, B * S, H * D, HID);
    gemm_wmma<false, false><<<dim3((KV * D) / 128, (B * S) / 128), blk, 0, stream>>>(
        x, wk, kf, B * S, KV * D, HID);
    gemm_wmma<false, true><<<dim3((KV * D) / 128, (B * S) / 128), blk, 0, stream>>>(
        x, wv, vbh, B * S, KV * D, HID);
    // RMSNorm + RoPE -> bf16 Q/K
    norm_rope<<<B * S, 128, 0, stream>>>(qf, kf, cosb, sinb, qnw, knw, qbh, kbh);
    // causal GQA flash attention -> bf16 attn output
    flash_attn<<<dim3(B * H, S / 64), 128, 0, stream>>>(qbh, kbh, vbh, attnb);
    // output projection -> fp32 d_out
    gemm_wmma<true, false><<<dim3(HID / 128, (B * S) / 128), blk, 0, stream>>>(
        attnb, wo, (float*)d_out, B * S, HID, H * D);
}
